// BasicVSRSA_34660386078899
// MI455X (gfx1250) — matrix operand order repaired and checked by reading, not compiled, not run on hardware
//
#include <hip/hip_runtime.h>
#include <hip/hip_bf16.h>
#include <math.h>

// ---------------------------------------------------------------------------
// BasicVSR-SA pipeline for gfx1250 (MI455X, wave32, WMMA).
// Conv3x3 / 1x1 convs run as implicit GEMM on v_wmma_f32_16x16x32_f16 with a
// two-stage LDS im2col so the K-loop is ds_load_b128 + global b128 + wmma.
// Raw tile staging uses the CDNA5 async global->LDS path when available.
// ---------------------------------------------------------------------------

typedef __attribute__((ext_vector_type(16))) _Float16 v16h;
typedef __attribute__((ext_vector_type(8)))  _Float16 v8h;
typedef __attribute__((ext_vector_type(8)))  float    v8f;

#define H48 48
#define W48 48
#define NPOS 2304            // 48*48
#define PADW 50
#define PADN 2500            // 50*50
#define HU 192
#define WU 192
#define NU 36864             // 192*192

constexpr int ACT_NONE = 0;
constexpr int ACT_RELU = 1;
constexpr int ACT_LRELU = 2;

// ---- CDNA5 async global->LDS (probe-confirmed: 4 args, typed int* params) --
#if defined(__HIP_DEVICE_COMPILE__) && defined(__gfx1250__) && \
    __has_builtin(__builtin_amdgcn_global_load_async_to_lds_b32)
#define HAVE_ASYNC_LDS 1
typedef __attribute__((address_space(1))) int gint_as1;
typedef __attribute__((address_space(3))) int lint_as3;
#else
#define HAVE_ASYNC_LDS 0
#endif

__device__ __forceinline__ void wait_async_lds() {
#if HAVE_ASYNC_LDS
#if __has_builtin(__builtin_amdgcn_s_wait_asynccnt)
  __builtin_amdgcn_s_wait_asynccnt(0);
#else
  asm volatile("s_wait_asynccnt 0" ::: "memory");
#endif
#endif
}

// ======================= weight fp32 -> fp16 (K-padded) =====================
__global__ void cvt_w_kernel(const float* __restrict__ w, _Float16* __restrict__ out,
                             int M, int Kin, int Kpad) {
  int idx = blockIdx.x * blockDim.x + threadIdx.x;
  if (idx >= M * Kpad) return;
  int m = idx / Kpad;
  int k = idx - m * Kpad;
  out[idx] = (k < Kin) ? (_Float16)w[m * Kin + k] : (_Float16)0.0f;
}

// ============ pack x_i(3ch) + feat(64ch or zeros) into padded f16 ===========
// Xpad layout: [72][50][50], channels 67..71 zero, 1-pixel zero border.
__global__ void pack_input72_kernel(const float* __restrict__ xi,
                                    const float* __restrict__ feat,
                                    _Float16* __restrict__ Xpad) {
  int idx = blockIdx.x * blockDim.x + threadIdx.x;
  if (idx >= 72 * PADN) return;
  int c = idx / PADN;
  int r = idx - c * PADN;
  int y = r / PADW;
  int x = r - y * PADW;
  float v = 0.0f;
  if (y >= 1 && y <= H48 && x >= 1 && x <= W48) {
    int p = (y - 1) * W48 + (x - 1);
    if (c < 3) v = xi[c * NPOS + p];
    else if (c < 67 && feat) v = feat[(c - 3) * NPOS + p];
  }
  Xpad[idx] = (_Float16)v;
}

// ===================== bilinear flow warp (zeros-at-border) =================
__global__ void flow_warp_kernel(const float* __restrict__ feat,
                                 const float* __restrict__ flow,
                                 float* __restrict__ out) {
  int p = blockIdx.x * blockDim.x + threadIdx.x;
  if (p >= NPOS) return;
  int y = p / W48, x = p - y * W48;
  float gx = (float)x + flow[p];
  float gy = (float)y + flow[NPOS + p];
  float x0f = floorf(gx), y0f = floorf(gy);
  float wx = gx - x0f, wy = gy - y0f;
  int ix[4], iy[4];
  float wt[4];
  float bx[2] = {x0f, x0f + 1.0f};
  float by[2] = {y0f, y0f + 1.0f};
  float fx[2] = {1.0f - wx, wx};
  float fy[2] = {1.0f - wy, wy};
#pragma unroll
  for (int t = 0; t < 4; ++t) {
    float tx = bx[t & 1], ty = by[t >> 1];
    float valid = (tx >= 0.0f && tx <= W48 - 1 && ty >= 0.0f && ty <= H48 - 1) ? 1.0f : 0.0f;
    ix[t] = (int)fminf(fmaxf(tx, 0.0f), (float)(W48 - 1));
    iy[t] = (int)fminf(fmaxf(ty, 0.0f), (float)(H48 - 1));
    wt[t] = fx[t & 1] * fy[t >> 1] * valid;
  }
  for (int c = 0; c < 64; ++c) {
    const float* fc = feat + c * NPOS;
    float v = 0.0f;
#pragma unroll
    for (int t = 0; t < 4; ++t) v += wt[t] * fc[iy[t] * W48 + ix[t]];
    out[c * NPOS + p] = v;
  }
}

__device__ __forceinline__ float apply_act(float v, int ACT) {
  if (ACT == ACT_RELU) return v > 0.0f ? v : 0.0f;
  if (ACT == ACT_LRELU) return v > 0.0f ? v : 0.1f * v;
  return v;
}

// =================== 3x3 conv as implicit GEMM on WMMA ======================
// Wk: [64][KPAD] f16 (K order = c*9 + dy*3 + dx), Xpad: [CPAD][50][50] f16.
// Block: 8 waves -> 64(M) x 32(N); grid = 2304/32 = 72 blocks.
// Stage 1: raw halo tile -> LDS (async global->LDS if available).
// Stage 2: cooperative im2col into Bs[32][KCHUNK] (incremental walk, no divs).
// Stage 3: K-loop: A = 2x global b128, B = 2x ds b128, v_wmma.
template <int KPAD, int CPAD, int ACT>
__global__ void __launch_bounds__(256)
conv3x3_wmma_kernel(const _Float16* __restrict__ Wk,
                    const float* __restrict__ bias,
                    const _Float16* __restrict__ Xpad,
                    const float* __restrict__ skip,
                    float* __restrict__ Yf32,
                    _Float16* __restrict__ Ypad) {
  constexpr int KCHUNK = 320;
  __shared__ __align__(16) _Float16 raw[CPAD * 200];     // 4 rows x 50 per channel
  __shared__ __align__(16) _Float16 Bs[32 * KCHUNK];     // im2col chunk, col-major K

  const int tid = threadIdx.x;
  const int blockN = blockIdx.x * 32;
  const int y0 = blockN / W48;

  // ---- stage raw halo tile: rows y0..y0+3, 50 cols, CPAD channels ----
  {
    const uint32_t* gsrc = (const uint32_t*)Xpad;
    uint32_t* sdst = (uint32_t*)raw;
    for (int w = tid; w < CPAD * 100; w += 256) {
      int c = w / 100;
      int r = w - c * 100;
      int row = r / 25;
      int cw = r - row * 25;
      int gy = y0 + row;
#if HAVE_ASYNC_LDS
      if (gy < PADW) {
        __builtin_amdgcn_global_load_async_to_lds_b32(
            (gint_as1*)(gsrc + c * 1250 + gy * 25 + cw),
            (lint_as3*)(sdst + w), 0, 0);
      } else {
        sdst[w] = 0u;
      }
#else
      sdst[w] = (gy < PADW) ? gsrc[c * 1250 + gy * 25 + cw] : 0u;
#endif
    }
    wait_async_lds();
  }
  __syncthreads();

  // ---- roles ----
  const int lane = tid & 31;
  const int wid = tid >> 5;
  const int mbase = (wid & 3) * 16;
  const int colL = (wid >> 2) * 16 + (lane & 15);  // 0..31 local output column
  const int colG = blockN + colL;
  const int cy = colG / W48;
  const int cx = colG - cy * W48;
  const int hi = lane >> 4;
  const int am = mbase + (lane & 15);
  const _Float16* wrow = Wk + am * KPAD + hi * 16;

  // im2col-builder role: column = tid&31, K-segment = tid>>5
  const int bcol = tid & 31;
  const int seg = tid >> 5;
  const int bcolG = blockN + bcol;
  const int bcy = bcolG / W48;
  const int bcx = bcolG - bcy * W48;
  const int blr = bcy - y0;                         // 0 or 1

  v8f acc = {};
#pragma unroll
  for (int kc = 0; kc < KPAD; kc += KCHUNK) {
    const int len = (KPAD - kc < KCHUNK) ? (KPAD - kc) : KCHUNK;  // 320 or 256

    // ---- build im2col chunk Bs[bcol][0..len) for k in [kc, kc+len) ----
    {
      const int kl = len >> 3;          // per-segment count (40 or 32)
      int k = kc + seg * kl;
      int c = k / 9;
      int rr = k - c * 9;
      int dy = rr / 3;
      int dx = rr - dy * 3;
      int idx = c * 200 + (blr + dy) * PADW + bcx + dx;
      int kk = seg * kl;
      _Float16* bdst = Bs + bcol * KCHUNK + kk;
      for (int t = 0; t < kl; ++t) {
        bdst[t] = raw[idx];
        if (dx < 2) { dx++; idx++; }
        else if (dy < 2) { dx = 0; dy++; idx += PADW - 2; }
        else { dx = 0; dy = 0; c++; idx += 200 - 2 * PADW - 2; }
      }
    }
    __syncthreads();

    // ---- consume chunk: fully unrolled WMMA K-loop ----
    const _Float16* bptr = Bs + colL * KCHUNK + hi * 16;
#pragma unroll
    for (int kk = 0; kk < len; kk += 32) {
      v8h alo = *(const v8h*)(wrow + kc + kk);       // K = kc+kk+hi*16 .. +7
      v8h ahi = *(const v8h*)(wrow + kc + kk + 8);   // K = kc+kk+hi*16+8 .. +15
      __builtin_prefetch(wrow + kc + kk + 64, 0, 1);
      v16h a;
#pragma unroll
      for (int j = 0; j < 8; ++j) { a[j] = alo[j]; a[8 + j] = ahi[j]; }
      v8h blo = *(const v8h*)(bptr + kk);            // 16 consecutive K per lane
      v8h bhi = *(const v8h*)(bptr + kk + 8);
      v16h b;
#pragma unroll
      for (int j = 0; j < 8; ++j) { b[j] = blo[j]; b[8 + j] = bhi[j]; }
      acc = __builtin_amdgcn_wmma_f32_16x16x32_f16(false, a, false, b,
                                                   (short)0, acc, false, false);
    }
    __syncthreads();
  }

#pragma unroll
  for (int r = 0; r < 8; ++r) {
    int m = mbase + r + hi * 8;
    float v = apply_act(acc[r] + bias[m], ACT);
    if (skip) v += skip[m * NPOS + colG];
    if (Yf32) Yf32[m * NPOS + colG] = v;
    if (Ypad) Ypad[m * PADN + (cy + 1) * PADW + (cx + 1)] = (_Float16)v;
  }
}

// ===================== 1x1 conv (plain GEMM) on WMMA ========================
// Y[m][n] = act(sum_k W[m][k] * X[k][n] + b[m]);  M = 64, grid = N/32.
// X chunk for the block's 32 columns staged into LDS column-major.
template <int KPAD, int ACT>
__global__ void __launch_bounds__(256)
gemm1x1_wmma_kernel(const _Float16* __restrict__ Wk,
                    const float* __restrict__ bias,
                    const _Float16* __restrict__ X,
                    float* __restrict__ Yf32,
                    _Float16* __restrict__ Yf16,
                    int N) {
  __shared__ __align__(16) _Float16 Bs[32 * KPAD];
  const int tid = threadIdx.x;
  const int blockN = blockIdx.x * 32;

  // stage: coalesced over columns for each k
  for (int idx = tid; idx < 32 * KPAD; idx += 256) {
    int k = idx >> 5;
    int cl = idx & 31;
    Bs[cl * KPAD + k] = X[(size_t)k * N + blockN + cl];
  }
  __syncthreads();

  const int lane = tid & 31;
  const int wid = tid >> 5;
  const int mbase = (wid & 3) * 16;
  const int colL = (wid >> 2) * 16 + (lane & 15);
  const int colG = blockN + colL;
  const int hi = lane >> 4;
  const int am = mbase + (lane & 15);
  const _Float16* wrow = Wk + am * KPAD + hi * 16;
  const _Float16* bptr = Bs + colL * KPAD + hi * 16;

  v8f acc = {};
#pragma unroll
  for (int k0 = 0; k0 < KPAD; k0 += 32) {
    v8h alo = *(const v8h*)(wrow + k0);
    v8h ahi = *(const v8h*)(wrow + k0 + 8);
    v16h a;
#pragma unroll
    for (int j = 0; j < 8; ++j) { a[j] = alo[j]; a[8 + j] = ahi[j]; }
    v8h blo = *(const v8h*)(bptr + k0);
    v8h bhi = *(const v8h*)(bptr + k0 + 8);
    v16h b;
#pragma unroll
    for (int j = 0; j < 8; ++j) { b[j] = blo[j]; b[8 + j] = bhi[j]; }
    acc = __builtin_amdgcn_wmma_f32_16x16x32_f16(false, a, false, b,
                                                 (short)0, acc, false, false);
  }
#pragma unroll
  for (int r = 0; r < 8; ++r) {
    int m = mbase + r + hi * 8;
    float v = apply_act(acc[r] + bias[m], ACT);
    if (Yf32) Yf32[(size_t)m * N + colG] = v;
    if (Yf16) Yf16[(size_t)m * N + colG] = (_Float16)v;
  }
}

// ============ pack concat(out_l, feat_prop) -> f16 [128][2304] ==============
__global__ void pack2_f16_kernel(const float* __restrict__ a,
                                 const float* __restrict__ b,
                                 _Float16* __restrict__ X) {
  int idx = blockIdx.x * blockDim.x + threadIdx.x;
  if (idx >= 128 * NPOS) return;
  float v = (idx < 64 * NPOS) ? a[idx] : b[idx - 64 * NPOS];
  X[idx] = (_Float16)v;
}

// =========== upsampler stage 1: emb1 = relu(W1 * inp + b1), K=4 =============
__global__ void up_head1_kernel(const float* __restrict__ w1,
                                const float* __restrict__ b1,
                                _Float16* __restrict__ emb1) {
  int idx = blockIdx.x * blockDim.x + threadIdx.x;
  if (idx >= 64 * NU) return;
  int o = idx / NU;
  int p = idx - o * NU;
  int i = p / WU, j = p - i * WU;
  float ah = ((float)i + 0.5f) * 0.25f;
  float aw = ((float)j + 0.5f) * 0.25f;
  float ch = ah - floorf(ah + 0.001f) - 0.5f;
  float cw = aw - floorf(aw + 0.001f) - 0.5f;
  float v = w1[o * 4 + 0] * 0.25f + w1[o * 4 + 1] * 0.25f +
            w1[o * 4 + 2] * ch + w1[o * 4 + 3] * cw + b1[o];
  emb1[idx] = (_Float16)(v > 0.0f ? v : 0.0f);
}

// === offset (2ch) + sigmoid routing (4ch) heads from emb2 (f16 [64][NU]) ====
__global__ void up_heads_kernel(const _Float16* __restrict__ emb2,
                                const float* __restrict__ off_w,
                                const float* __restrict__ off_b,
                                const float* __restrict__ route_w,
                                const float* __restrict__ route_b,
                                float* __restrict__ offset,
                                float* __restrict__ rw) {
  int p = blockIdx.x * blockDim.x + threadIdx.x;
  if (p >= NU) return;
  float e[64];
#pragma unroll
  for (int k = 0; k < 64; ++k) e[k] = (float)emb2[(size_t)k * NU + p];
#pragma unroll
  for (int o = 0; o < 2; ++o) {
    float v = off_b[o];
    for (int k = 0; k < 64; ++k) v += off_w[o * 64 + k] * e[k];
    offset[o * NU + p] = v;
  }
#pragma unroll
  for (int o = 0; o < 4; ++o) {
    float v = route_b[o];
    for (int k = 0; k < 64; ++k) v += route_w[o * 64 + k] * e[k];
    rw[o * NU + p] = 1.0f / (1.0f + expf(-v));
  }
}

// ========== bilinear sample fused feature at offset-adjusted grid ===========
__global__ void fea0_kernel(const float* __restrict__ fused,
                            const float* __restrict__ offset,
                            float* __restrict__ fea0) {
  int p = blockIdx.x * blockDim.x + threadIdx.x;
  if (p >= NU) return;
  int i = p / WU, j = p - i * WU;
  float gx = ((float)j + 0.5f) * 0.25f - 0.5f + offset[p];
  float gy = ((float)i + 0.5f) * 0.25f - 0.5f + offset[NU + p];
  float x0f = floorf(gx), y0f = floorf(gy);
  float wx = gx - x0f, wy = gy - y0f;
  int ix[4], iy[4];
  float wt[4];
  float bx[2] = {x0f, x0f + 1.0f};
  float by[2] = {y0f, y0f + 1.0f};
  float fx[2] = {1.0f - wx, wx};
  float fy[2] = {1.0f - wy, wy};
#pragma unroll
  for (int t = 0; t < 4; ++t) {
    float tx = bx[t & 1], ty = by[t >> 1];
    float valid = (tx >= 0.0f && tx <= W48 - 1 && ty >= 0.0f && ty <= H48 - 1) ? 1.0f : 0.0f;
    ix[t] = (int)fminf(fmaxf(tx, 0.0f), (float)(W48 - 1));
    iy[t] = (int)fminf(fmaxf(ty, 0.0f), (float)(H48 - 1));
    wt[t] = fx[t & 1] * fy[t >> 1] * valid;
  }
  for (int c = 0; c < 64; ++c) {
    const float* fc = fused + c * NPOS;
    float v = 0.0f;
#pragma unroll
    for (int t = 0; t < 4; ++t) v += wt[t] * fc[iy[t] * W48 + ix[t]];
    fea0[(size_t)c * NU + p] = v;
  }
}

// ==================== 4-expert low-rank MoE + residual ======================
__global__ void moe_kernel(const float* __restrict__ rw,
                           const float* __restrict__ wc,
                           const float* __restrict__ we,
                           const float* __restrict__ fea0,
                           float* __restrict__ out) {
  int p = blockIdx.x * blockDim.x + threadIdx.x;
  if (p >= NU) return;
  float r[4];
#pragma unroll
  for (int e = 0; e < 4; ++e) r[e] = rw[e * NU + p];
  float f0[64];
  for (int c = 0; c < 64; ++c) f0[c] = fea0[(size_t)c * NU + p];
  float mid[8];
#pragma unroll
  for (int k = 0; k < 8; ++k) {
    float acc = 0.0f;
#pragma unroll
    for (int e = 0; e < 4; ++e) {
      const float* wck = wc + e * 512 + k * 64;
      float d = 0.0f;
      for (int c = 0; c < 64; ++c) d += wck[c] * f0[c];
      acc += r[e] * d;
    }
    mid[k] = acc;
  }
  for (int c = 0; c < 64; ++c) {
    float acc = 0.0f;
#pragma unroll
    for (int e = 0; e < 4; ++e) {
      const float* wec = we + e * 512 + c * 8;
      float d = 0.0f;
#pragma unroll
      for (int k = 0; k < 8; ++k) d += wec[k] * mid[k];
      acc += r[e] * d;
    }
    out[(size_t)c * NU + p] = acc + f0[c];
  }
}

// ===========================================================================
//                                host side
// ===========================================================================
namespace {
struct TrunkW {
  const float *in_w, *in_b, *bw1, *bb1, *bw2, *bb2;
};

struct TrunkF16 {
  _Float16 *win;   // [64][640]
  _Float16 *wb1;   // [15*64][576]
  _Float16 *wb2;   // [15*64][576]
};

static inline size_t alloc(size_t& off, size_t bytes) {
  size_t a = (off + 255) & ~(size_t)255;
  off = a + bytes;
  return a;
}

static void run_trunk(const TrunkF16& w, const TrunkW& tw,
                      const _Float16* Xpad72, _Float16* padA, _Float16* padB,
                      float* Fbuf, float* finalF, hipStream_t stream) {
  dim3 grid(NPOS / 32), blk(256);
  // input conv (67->64), leaky relu; K padded 603->640, Cpad=72
  conv3x3_wmma_kernel<640, 72, ACT_LRELU><<<grid, blk, 0, stream>>>(
      w.win, tw.in_b, Xpad72, nullptr, Fbuf, padA);
  for (int b = 0; b < 15; ++b) {
    conv3x3_wmma_kernel<576, 64, ACT_RELU><<<grid, blk, 0, stream>>>(
        w.wb1 + (size_t)b * 64 * 576, tw.bb1 + b * 64, padA, nullptr,
        nullptr, padB);
    float* yout = (b == 14) ? finalF : Fbuf;
    _Float16* ypad = (b == 14) ? nullptr : padA;
    conv3x3_wmma_kernel<576, 64, ACT_NONE><<<grid, blk, 0, stream>>>(
        w.wb2 + (size_t)b * 64 * 576, tw.bb2 + b * 64, padB, Fbuf,
        yout, ypad);
  }
}
}  // namespace

extern "C" void kernel_launch(void* const* d_in, const int* in_sizes, int n_in,
                              void* d_out, int out_size, void* d_ws, size_t ws_size,
                              hipStream_t stream) {
  (void)in_sizes; (void)n_in; (void)out_size; (void)ws_size;
  // ---- inputs (setup_inputs() dict order; params flattened in insertion order)
  const float* x  = (const float*)d_in[0];   // (1,4,3,48,48)
  const float* ff = (const float*)d_in[1];   // (1,3,2,48,48)
  const float* fb = (const float*)d_in[2];   // (1,3,2,48,48)
  TrunkW bwd { (const float*)d_in[3], (const float*)d_in[4], (const float*)d_in[5],
               (const float*)d_in[6], (const float*)d_in[7], (const float*)d_in[8] };
  TrunkW fwd { (const float*)d_in[9], (const float*)d_in[10], (const float*)d_in[11],
               (const float*)d_in[12], (const float*)d_in[13], (const float*)d_in[14] };
  const float* fusion_w = (const float*)d_in[15];  // (64,128,1,1)
  const float* fusion_b = (const float*)d_in[16];
  const float* body_w1  = (const float*)d_in[17];  // (64,4,1,1)
  const float* body_b1  = (const float*)d_in[18];
  const float* body_w2  = (const float*)d_in[19];  // (64,64,1,1)
  const float* body_b2  = (const float*)d_in[20];
  const float* route_w  = (const float*)d_in[21];  // (4,64,1,1)
  const float* route_b  = (const float*)d_in[22];
  const float* off_w    = (const float*)d_in[23];  // (2,64,1,1)
  const float* off_b    = (const float*)d_in[24];
  const float* wc       = (const float*)d_in[25];  // (4,8,64)
  const float* we       = (const float*)d_in[26];  // (4,64,8)
  float* out = (float*)d_out;                      // (1,4,64,192,192)

  // ---- workspace carve-up
  char* ws = (char*)d_ws;
  size_t off = 0;
  float*    warp_buf  = (float*)   (ws + alloc(off, 64 * NPOS * 4));
  float*    Fbuf      = (float*)   (ws + alloc(off, 64 * NPOS * 4));
  float*    fwd_cur   = (float*)   (ws + alloc(off, 64 * NPOS * 4));
  float*    out_l     = (float*)   (ws + alloc(off, 4 * 64 * NPOS * 4));
  float*    fused_buf = (float*)   (ws + alloc(off, 64 * NPOS * 4));
  _Float16* Xpad72    = (_Float16*)(ws + alloc(off, 72 * PADN * 2));
  _Float16* padA      = (_Float16*)(ws + alloc(off, 64 * PADN * 2));
  _Float16* padB      = (_Float16*)(ws + alloc(off, 64 * PADN * 2));
  _Float16* Xfused    = (_Float16*)(ws + alloc(off, 128 * NPOS * 2));
  TrunkF16 wbwd, wfwd;
  wbwd.win = (_Float16*)(ws + alloc(off, 64 * 640 * 2));
  wbwd.wb1 = (_Float16*)(ws + alloc(off, (size_t)960 * 576 * 2));
  wbwd.wb2 = (_Float16*)(ws + alloc(off, (size_t)960 * 576 * 2));
  wfwd.win = (_Float16*)(ws + alloc(off, 64 * 640 * 2));
  wfwd.wb1 = (_Float16*)(ws + alloc(off, (size_t)960 * 576 * 2));
  wfwd.wb2 = (_Float16*)(ws + alloc(off, (size_t)960 * 576 * 2));
  _Float16* Wfus   = (_Float16*)(ws + alloc(off, 64 * 128 * 2));
  _Float16* Wbody2 = (_Float16*)(ws + alloc(off, 64 * 64 * 2));
  _Float16* emb1   = (_Float16*)(ws + alloc(off, (size_t)64 * NU * 2));
  _Float16* emb2   = (_Float16*)(ws + alloc(off, (size_t)64 * NU * 2));
  float*    offbuf = (float*)   (ws + alloc(off, (size_t)2 * NU * 4));
  float*    rwbuf  = (float*)   (ws + alloc(off, (size_t)4 * NU * 4));
  float*    fea0   = (float*)   (ws + alloc(off, (size_t)64 * NU * 4));

  dim3 b256(256);
  auto g = [](int n) { return dim3((n + 255) / 256); };

  // ---- per-launch weight conversion fp32 -> fp16
  cvt_w_kernel<<<g(64 * 640), b256, 0, stream>>>(bwd.in_w, wbwd.win, 64, 603, 640);
  cvt_w_kernel<<<g(960 * 576), b256, 0, stream>>>(bwd.bw1, wbwd.wb1, 960, 576, 576);
  cvt_w_kernel<<<g(960 * 576), b256, 0, stream>>>(bwd.bw2, wbwd.wb2, 960, 576, 576);
  cvt_w_kernel<<<g(64 * 640), b256, 0, stream>>>(fwd.in_w, wfwd.win, 64, 603, 640);
  cvt_w_kernel<<<g(960 * 576), b256, 0, stream>>>(fwd.bw1, wfwd.wb1, 960, 576, 576);
  cvt_w_kernel<<<g(960 * 576), b256, 0, stream>>>(fwd.bw2, wfwd.wb2, 960, 576, 576);
  cvt_w_kernel<<<g(64 * 128), b256, 0, stream>>>(fusion_w, Wfus, 64, 128, 128);
  cvt_w_kernel<<<g(64 * 64), b256, 0, stream>>>(body_w2, Wbody2, 64, 64, 64);

  // zero padded f16 activation buffers so the 1-px border stays zero
  (void)hipMemsetAsync(padA, 0, 64 * PADN * 2, stream);
  (void)hipMemsetAsync(padB, 0, 64 * PADN * 2, stream);

  // ---- backward propagation (i = 3..0)
  for (int i = 3; i >= 0; --i) {
    const float* featsrc = nullptr;
    if (i < 3) {
      flow_warp_kernel<<<g(NPOS), b256, 0, stream>>>(
          out_l + (size_t)(i + 1) * 64 * NPOS, fb + (size_t)i * 2 * NPOS, warp_buf);
      featsrc = warp_buf;
    }
    pack_input72_kernel<<<g(72 * PADN), b256, 0, stream>>>(
        x + (size_t)i * 3 * NPOS, featsrc, Xpad72);
    run_trunk(wbwd, bwd, Xpad72, padA, padB, Fbuf,
              out_l + (size_t)i * 64 * NPOS, stream);
  }

  // ---- forward propagation + fusion + upsample (i = 0..3)
  for (int i = 0; i < 4; ++i) {
    const float* featsrc = nullptr;
    if (i > 0) {
      flow_warp_kernel<<<g(NPOS), b256, 0, stream>>>(
          fwd_cur, ff + (size_t)(i - 1) * 2 * NPOS, warp_buf);
      featsrc = warp_buf;
    }
    pack_input72_kernel<<<g(72 * PADN), b256, 0, stream>>>(
        x + (size_t)i * 3 * NPOS, featsrc, Xpad72);
    run_trunk(wfwd, fwd, Xpad72, padA, padB, Fbuf, fwd_cur, stream);

    // fusion: lrelu(1x1 conv of concat(out_l[i], feat_prop))
    pack2_f16_kernel<<<g(128 * NPOS), b256, 0, stream>>>(
        out_l + (size_t)i * 64 * NPOS, fwd_cur, Xfused);
    gemm1x1_wmma_kernel<128, ACT_LRELU><<<dim3(NPOS / 32), b256, 0, stream>>>(
        Wfus, fusion_b, Xfused, fused_buf, nullptr, NPOS);

    // upsampler
    up_head1_kernel<<<g(64 * NU), b256, 0, stream>>>(body_w1, body_b1, emb1);
    gemm1x1_wmma_kernel<64, ACT_RELU><<<dim3(NU / 32), b256, 0, stream>>>(
        Wbody2, body_b2, emb1, nullptr, emb2, NU);
    up_heads_kernel<<<g(NU), b256, 0, stream>>>(
        emb2, off_w, off_b, route_w, route_b, offbuf, rwbuf);
    fea0_kernel<<<g(NU), b256, 0, stream>>>(fused_buf, offbuf, fea0);
    moe_kernel<<<g(NU), b256, 0, stream>>>(
        rwbuf, wc, we, fea0, out + (size_t)i * 64 * NU);
  }
}
